// MultiHeadAttention_72327249264996
// MI455X (gfx1250) — matrix operand order repaired and checked by reading, not compiled, not run on hardware
//
#include <hip/hip_runtime.h>

// ---------------------------------------------------------------------------
// MI455X (gfx1250, wave32) multi-head attention:
//   GEMM(QKV, bf16 WMMA) -> flash attention (bf16 WMMA, online softmax)
//   -> GEMM(out proj, bf16 WMMA).  f32 accumulate throughout.
//   K/V tiles and bf16 A tiles staged to LDS with the Tensor Data Mover
//   (tensor_load_to_lds + s_wait_tensorcnt), padded rows to dodge LDS banks.
// ---------------------------------------------------------------------------

#define S_LEN  2048
#define DMODEL 1024
#define NH     16
#define HD     64

typedef __bf16 bf16_t;
typedef bf16_t v16bf __attribute__((ext_vector_type(16)));
typedef float  v8f   __attribute__((ext_vector_type(8)));
typedef unsigned v4u_t __attribute__((ext_vector_type(4)));
typedef int      v8i_t __attribute__((ext_vector_type(8)));
typedef int      v4i_t __attribute__((ext_vector_type(4)));

union Frag { v16bf v; unsigned u[8]; };

#if defined(__has_builtin)
#if __has_builtin(__builtin_amdgcn_tensor_load_to_lds)
#define HAVE_TDM 1
#endif
#endif

__device__ __forceinline__ unsigned short f2bf(float f) {
    unsigned u = __float_as_uint(f);
    u += 0x7FFFu + ((u >> 16) & 1u);          // round-to-nearest-even
    return (unsigned short)(u >> 16);
}

// 16x32 bf16 A-matrix layout: lane L holds M=L%16; half h=L/16.
// VGPR v packs K pair: v<4 -> K=2v+8h ; v>=4 -> K=16+2(v-4)+8h
__device__ __forceinline__ int a_koff(int v, int h) {
    return ((v & 4) << 2) + ((v & 3) << 1) + (h << 3);
}
// 32x16 bf16 B-matrix layout: lane L holds N=L%16; half h=L/16.
// VGPR v packs K pair: v<4 -> K=2v+8h ; v>=4 -> K=16+2(v-4)+8h
__device__ __forceinline__ int b_koff(int v, int h) {
    return ((v & 4) << 2) + ((v & 3) << 1) + (h << 3);
}

// Low 32 bits of a generic pointer to __shared__ == LDS byte offset
// (LDS aperture maps addr[31:0] directly; see ISA 10.2).
__device__ __forceinline__ unsigned lds_off(const void* p) {
    return (unsigned)(unsigned long long)p;
}

// ---------------------------------------------------------------------------
// TDM: DMA a 2D bf16 tile [rows = tile_d1][cols = tile_d0] from global
// (row stride = stride_elems) into LDS at lds_addr, inserting pad_amount
// (encoded) DWORDs of padding every 2^(pad_interval_enc+1) DWORDs.
// Issued once per wave; tracked with TENSORcnt.
// ---------------------------------------------------------------------------
__device__ __forceinline__ void tdm_load_2d_bf16(unsigned lds_addr, const void* gaddr,
                                                 int tile_d0, int tile_d1,
                                                 int stride_elems,
                                                 int pad_interval_enc, int pad_amount_enc)
{
#if defined(HAVE_TDM)
    unsigned long long ga = (unsigned long long)gaddr;
    v4u_t g0;
    g0.x = 1u;                                  // count=1, user descriptor
    g0.y = lds_addr;                            // lds_addr [63:32]
    g0.z = (unsigned)ga;                        // global_addr low   [95:64]
    g0.w = (unsigned)(ga >> 32) | (2u << 30);   // global_addr high | type=2
    v8i_t g1;
    g1[0] = (1 << 16)                           // data_size = 2 bytes
          | (1 << 20)                           // pad_enable
          | (pad_interval_enc << 22)
          | (pad_amount_enc << 25);
    g1[1] = (tile_d0 & 0xffff) << 16;                                  // tensor_dim0 lo
    g1[2] = ((tile_d0 >> 16) & 0xffff) | ((tile_d1 & 0xffff) << 16);   // d0 hi | tensor_dim1 lo
    g1[3] = ((tile_d1 >> 16) & 0xffff) | ((tile_d0 & 0xffff) << 16);   // d1 hi | tile_dim0
    g1[4] = (tile_d1 & 0xffff);                                        // tile_dim1 (tile_dim2=0)
    g1[5] = stride_elems;                                              // tensor_dim0_stride lo32
    g1[6] = 0;
    g1[7] = 0;
    v4i_t z4 = {0, 0, 0, 0};
    v8i_t z8 = {0, 0, 0, 0, 0, 0, 0, 0};
    // 6-arg toolchain signature: (g0, g1, g2, g3, extra_group, cpol)
    __builtin_amdgcn_tensor_load_to_lds(g0, g1, z4, z4, z8, 0);
#else
    (void)lds_addr; (void)gaddr; (void)tile_d0; (void)tile_d1;
    (void)stride_elems; (void)pad_interval_enc; (void)pad_amount_enc;
#endif
}

__device__ __forceinline__ void tdm_wait() {
#if defined(HAVE_TDM)
    __builtin_amdgcn_s_wait_tensorcnt(0);
#endif
}

// Fallback async copy (ASYNCcnt path) for toolchains without the TDM builtin.
__device__ __forceinline__ void async_ld_b128(unsigned lds_byte, const void* g) {
    asm volatile("global_load_async_to_lds_b128 %0, %1, off"
                 :: "v"(lds_byte), "v"(g) : "memory");
}
__device__ __forceinline__ void async_wait() {
    asm volatile("s_wait_asynccnt 0x0" ::: "memory");
}

// ---------------------------------------------------------------------------
// Tiled GEMM:  C[M,N] = A[M,K] * B[K,N] + bias
//   block tile 128x128, BK=32 (one WMMA k-step), 8 waves -> 64x32 per wave.
//   EPI==0: scatter into Q (x0.125), K, V head-major bf16 buffers.
//   EPI==1: write f32 to Out.
// ---------------------------------------------------------------------------
template<bool A_BF16, int EPI>
__global__ __launch_bounds__(256)
void gemm_wmma(const void* __restrict__ Av, const float* __restrict__ Bmat,
               const float* __restrict__ bias,
               unsigned short* __restrict__ Qd, unsigned short* __restrict__ Kd,
               unsigned short* __restrict__ Vd, float* __restrict__ Out,
               int M, int N, int Kdim)
{
    __shared__ unsigned short As[128 * 40];   // [m][k]  64B row + 16B pad
    __shared__ unsigned short Bs[128 * 40];   // [n][k]  transposed, stride 40

    const int tid   = threadIdx.x;
    const int lane  = tid & 31;
    const int wid   = tid >> 5;
    const int waveM = wid >> 2;               // 0..1 -> 64 rows each
    const int waveN = wid & 3;                // 0..3 -> 32 cols each
    const int half  = lane >> 4;
    const int nl    = lane & 15;

    const int bm = blockIdx.y * 128;
    const int bn = blockIdx.x * 128;

    v8f acc[4][2];
#pragma unroll
    for (int i = 0; i < 4; ++i)
#pragma unroll
        for (int j = 0; j < 2; ++j) acc[i][j] = (v8f)0.f;

    for (int k0 = 0; k0 < Kdim; k0 += 32) {
        // ---- stage A tile (128 x 32) as bf16 ----
        if (A_BF16) {
            const unsigned short* A = (const unsigned short*)Av;
#if defined(HAVE_TDM)
            if (wid == 0) {
                // TDM: 128 rows x 32 cols, row = 64B = 16 dwords -> pad_interval enc 3,
                // pad 16B = 4 dwords -> enc 3; LDS row stride = 80B (stride 40 halves).
                tdm_load_2d_bf16(lds_off(&As[0]),
                                 A + (size_t)bm * Kdim + k0,
                                 32, 128, Kdim, 3, 3);
                tdm_wait();
            }
#else
#pragma unroll
            for (int it = 0; it < 2; ++it) {
                int lin = it * 2048 + tid * 8;
                int r = lin >> 5, c = lin & 31;
                async_ld_b128(lds_off(&As[r * 40 + c]),
                              A + (size_t)(bm + r) * Kdim + k0 + c);
            }
            async_wait();
#endif
        } else {
            const float* A = (const float*)Av;
#pragma unroll
            for (int it = 0; it < 4; ++it) {
                int lin = it * 1024 + tid * 4;
                int r = lin >> 5, c = lin & 31;
                float4 d = *(const float4*)(A + (size_t)(bm + r) * Kdim + k0 + c);
                uint2 p;
                p.x = (unsigned)f2bf(d.x) | ((unsigned)f2bf(d.y) << 16);
                p.y = (unsigned)f2bf(d.z) | ((unsigned)f2bf(d.w) << 16);
                *(uint2*)(&As[r * 40 + c]) = p;
            }
        }
        // ---- stage B tile (32 x 128) f32->bf16, transposed to [n][k] ----
#pragma unroll
        for (int it = 0; it < 4; ++it) {
            int lin = it * 1024 + tid * 4;
            int kr = lin >> 7, c = lin & 127;
            float4 d = *(const float4*)(Bmat + (size_t)(k0 + kr) * N + bn + c);
            Bs[(c + 0) * 40 + kr] = f2bf(d.x);
            Bs[(c + 1) * 40 + kr] = f2bf(d.y);
            Bs[(c + 2) * 40 + kr] = f2bf(d.z);
            Bs[(c + 3) * 40 + kr] = f2bf(d.w);
        }
        if (k0 + 32 < Kdim) {   // prefetch next B k-tile (global_prefetch_b8)
            __builtin_prefetch(Bmat + (size_t)(k0 + 32 + (tid >> 5)) * N + bn + (tid & 31) * 4, 0, 1);
        }
        __syncthreads();

        Frag afr[4], bfr[2];
#pragma unroll
        for (int mt = 0; mt < 4; ++mt) {
            const unsigned short* ap = &As[(waveM * 64 + mt * 16 + nl) * 40];
#pragma unroll
            for (int v = 0; v < 8; ++v)
                afr[mt].u[v] = *(const unsigned*)(ap + a_koff(v, half));
        }
#pragma unroll
        for (int nt = 0; nt < 2; ++nt) {
            const unsigned short* bp = &Bs[(waveN * 32 + nt * 16 + nl) * 40];
#pragma unroll
            for (int v = 0; v < 8; ++v)
                bfr[nt].u[v] = *(const unsigned*)(bp + b_koff(v, half));
        }
#pragma unroll
        for (int mt = 0; mt < 4; ++mt)
#pragma unroll
            for (int nt = 0; nt < 2; ++nt)
                acc[mt][nt] = __builtin_amdgcn_wmma_f32_16x16x32_bf16(
                    false, afr[mt].v, false, bfr[nt].v,
                    (short)0, acc[mt][nt], false, false);
        __syncthreads();
    }

    // ---- epilogue ----
#pragma unroll
    for (int mt = 0; mt < 4; ++mt) {
#pragma unroll
        for (int nt = 0; nt < 2; ++nt) {
            int gn = bn + waveN * 32 + nt * 16 + nl;
            float bv = bias[gn];
#pragma unroll
            for (int r = 0; r < 8; ++r) {
                int gm = bm + waveM * 64 + mt * 16 + r + half * 8;
                float val = acc[mt][nt][r] + bv;
                if (EPI == 0) {
                    int h = gn / (3 * HD);
                    int j = gn % (3 * HD);
                    int b = gm >> 11, s = gm & (S_LEN - 1);
                    size_t base = ((size_t)(b * NH + h) * S_LEN + s) * HD;
                    if (j < HD)            Qd[base + j]          = f2bf(val * 0.125f); // fold 1/sqrt(64)
                    else if (j < 2 * HD)   Kd[base + j - HD]     = f2bf(val);
                    else                   Vd[base + j - 2 * HD] = f2bf(val);
                } else {
                    Out[(size_t)gm * N + gn] = val;
                }
            }
        }
    }
}

// ---------------------------------------------------------------------------
// Flash attention per (b,h): block = 128 query rows (8 waves x 16 rows),
// loop over 128-key blocks with online softmax. Q pre-scaled by 0.125.
// K and V tiles are DMA'd into LDS by the Tensor Data Mover.
// ---------------------------------------------------------------------------
__global__ __launch_bounds__(256)
void attn_wmma(const unsigned short* __restrict__ Q,
               const unsigned short* __restrict__ K,
               const unsigned short* __restrict__ V,
               unsigned short* __restrict__ O)
{
    __shared__ unsigned short Ks[128 * 72];     // [key][hd] 128B row + 16B pad
    __shared__ unsigned short Vs[128 * 72];     // [key][hd] same layout
    __shared__ unsigned short Ps[8][16 * 36];   // per-wave P staging, stride 36

    const int tid  = threadIdx.x;
    const int lane = tid & 31;
    const int w    = tid >> 5;
    const int half = lane >> 4;
    const int nl   = lane & 15;

    const size_t hoff = (size_t)blockIdx.y * S_LEN * HD;
    const unsigned short* Qh = Q + hoff;
    const unsigned short* Kh = K + hoff;
    const unsigned short* Vh = V + hoff;
    unsigned short*       Oh = O + hoff;

    const int q0 = blockIdx.x * 128 + w * 16;

    // Q fragments for this wave's 16 rows (two k-steps over HD=64)
    Frag qf[2];
    {
        const unsigned short* qp = Qh + (size_t)(q0 + nl) * HD;
#pragma unroll
        for (int ks = 0; ks < 2; ++ks)
#pragma unroll
            for (int v = 0; v < 8; ++v)
                qf[ks].u[v] = *(const unsigned*)(qp + ks * 32 + a_koff(v, half));
    }

    v8f oacc[4];
#pragma unroll
    for (int i = 0; i < 4; ++i) oacc[i] = (v8f)0.f;
    float mrow[8], lrow[8];
#pragma unroll
    for (int r = 0; r < 8; ++r) { mrow[r] = -1e30f; lrow[r] = 0.f; }

    for (int kb = 0; kb < S_LEN / 128; ++kb) {
        const int kbase = kb * 128;
        // ---- stage K and V tiles: 128 keys x 64 hd, row 128B = 32 dwords
        //      -> pad_interval enc 4, pad 4 dwords enc 3; LDS stride 144B. ----
#if defined(HAVE_TDM)
        if (w == 0) {
            tdm_load_2d_bf16(lds_off(&Ks[0]), Kh + (size_t)kbase * HD,
                             HD, 128, HD, 4, 3);
            tdm_wait();
        } else if (w == 1) {
            tdm_load_2d_bf16(lds_off(&Vs[0]), Vh + (size_t)kbase * HD,
                             HD, 128, HD, 4, 3);
            tdm_wait();
        }
#else
#pragma unroll
        for (int it = 0; it < 2; ++it) {
            int lin = it * 2048 + tid * 8;
            int key = lin >> 6, hd = lin & 63;
            async_ld_b128(lds_off(&Ks[key * 72 + hd]),
                          Kh + (size_t)(kbase + key) * HD + hd);
            async_ld_b128(lds_off(&Vs[key * 72 + hd]),
                          Vh + (size_t)(kbase + key) * HD + hd);
        }
        async_wait();
#endif
        __syncthreads();

        // ---- S = Q K^T : 16 x 128 in 8 accumulator tiles ----
        v8f sacc[8];
#pragma unroll
        for (int nt = 0; nt < 8; ++nt) sacc[nt] = (v8f)0.f;
#pragma unroll
        for (int nt = 0; nt < 8; ++nt) {
#pragma unroll
            for (int ks = 0; ks < 2; ++ks) {
                Frag kfr;
                const unsigned short* bp = &Ks[(nt * 16 + nl) * 72 + ks * 32];
#pragma unroll
                for (int v = 0; v < 8; ++v)
                    kfr.u[v] = *(const unsigned*)(bp + b_koff(v, half));
                sacc[nt] = __builtin_amdgcn_wmma_f32_16x16x32_bf16(
                    false, qf[ks].v, false, kfr.v, (short)0, sacc[nt], false, false);
            }
        }

        // ---- online softmax (row r lives in a 16-lane half: xor-reduce) ----
#pragma unroll
        for (int r = 0; r < 8; ++r) {
            float mx = sacc[0][r];
#pragma unroll
            for (int nt = 1; nt < 8; ++nt) mx = fmaxf(mx, sacc[nt][r]);
#pragma unroll
            for (int d = 1; d < 16; d <<= 1) mx = fmaxf(mx, __shfl_xor(mx, d, 32));
            float mnew  = fmaxf(mrow[r], mx);
            float scale = __expf(mrow[r] - mnew);
            float rs = 0.f;
#pragma unroll
            for (int nt = 0; nt < 8; ++nt) {
                float p = __expf(sacc[nt][r] - mnew);
                sacc[nt][r] = p;
                rs += p;
            }
#pragma unroll
            for (int d = 1; d < 16; d <<= 1) rs += __shfl_xor(rs, d, 32);
            lrow[r] = lrow[r] * scale + rs;
            mrow[r] = mnew;
#pragma unroll
            for (int ht = 0; ht < 4; ++ht) oacc[ht][r] *= scale;
        }

        // ---- O += P V : stage P 16x32 chunks through wave-private LDS ----
        unsigned short* Pw = Ps[w];
#pragma unroll
        for (int kc = 0; kc < 4; ++kc) {
#pragma unroll
            for (int t = 0; t < 2; ++t) {
                int nt = kc * 2 + t;
#pragma unroll
                for (int r = 0; r < 8; ++r)
                    Pw[(r + half * 8) * 36 + t * 16 + nl] = f2bf(sacc[nt][r]);
            }
            asm volatile("s_wait_dscnt 0x0" ::: "memory");   // LDS RAW, same wave
            Frag pf;
            const unsigned short* pp = &Pw[nl * 36];
#pragma unroll
            for (int v = 0; v < 8; ++v)
                pf.u[v] = *(const unsigned*)(pp + a_koff(v, half));
#pragma unroll
            for (int ht = 0; ht < 4; ++ht) {
                // B-frag from row-major V: gather key-pairs (two u16 reads, pack)
                Frag vfr;
#pragma unroll
                for (int v = 0; v < 8; ++v) {
                    int krow = kc * 32 + b_koff(v, half);
                    unsigned lo = Vs[krow * 72 + ht * 16 + nl];
                    unsigned hi = Vs[(krow + 1) * 72 + ht * 16 + nl];
                    vfr.u[v] = lo | (hi << 16);
                }
                oacc[ht] = __builtin_amdgcn_wmma_f32_16x16x32_bf16(
                    false, pf.v, false, vfr.v, (short)0, oacc[ht], false, false);
            }
        }
        __syncthreads();
    }

    // ---- normalize and write O (bf16, [b,h,s,hd] contiguous) ----
#pragma unroll
    for (int ht = 0; ht < 4; ++ht) {
#pragma unroll
        for (int r = 0; r < 8; ++r) {
            float val = oacc[ht][r] / lrow[r];
            int row = q0 + r + half * 8;
            Oh[(size_t)row * HD + ht * 16 + nl] = f2bf(val);
        }
    }
}

// ---------------------------------------------------------------------------
extern "C" void kernel_launch(void* const* d_in, const int* in_sizes, int n_in,
                              void* d_out, int out_size, void* d_ws, size_t ws_size,
                              hipStream_t stream)
{
    (void)in_sizes; (void)n_in; (void)out_size; (void)ws_size;
    const float* x    = (const float*)d_in[0];
    const float* Wqkv = (const float*)d_in[1];
    const float* bqkv = (const float*)d_in[2];
    const float* Wo   = (const float*)d_in[3];
    const float* bo   = (const float*)d_in[4];
    float* out = (float*)d_out;

    const int B = 4;
    const size_t headElems = (size_t)B * NH * S_LEN * HD;   // 8M elems per buffer
    unsigned short* Qws = (unsigned short*)d_ws;            // bf16 bits
    unsigned short* Kws = Qws + headElems;
    unsigned short* Vws = Kws + headElems;
    unsigned short* Ows = Vws + headElems;                  // total 64 MB

    const int M = B * S_LEN;                                // 8192

    // 1) fused QKV projection + head scatter (+ fold 1/sqrt(HD) into Q)
    dim3 g1(3 * DMODEL / 128, M / 128);
    gemm_wmma<false, 0><<<g1, 256, 0, stream>>>(
        (const void*)x, Wqkv, bqkv, Qws, Kws, Vws, nullptr, M, 3 * DMODEL, DMODEL);

    // 2) flash attention per (b,h)
    dim3 g2(S_LEN / 128, B * NH);
    attn_wmma<<<g2, 256, 0, stream>>>(Qws, Kws, Vws, Ows);

    // 3) output projection: reference reshapes [B,H,S,HD]->[B,S,D] without
    //    permuting, so Ows is directly the row-major A matrix here.
    dim3 g3(DMODEL / 128, M / 128);
    gemm_wmma<true, 1><<<g3, 256, 0, stream>>>(
        (const void*)Ows, Wo, bo, nullptr, nullptr, nullptr, out, M, DMODEL, DMODEL);
}
